// FeatureExtraction_77189152244217
// MI455X (gfx1250) — matrix operand order repaired and checked by reading, not compiled, not run on hardware
//
#include <hip/hip_runtime.h>
#include <hip/hip_bf16.h>

// ---------------------------------------------------------------------------
// PointNet++ MSG forward for MI455X (gfx1250, wave32).
// GEMMs: v_wmma_f32_16x16x32_f16 (f16 in, f32 accum; BN re-normalizes every
// layer so f16 intermediates are safe). Workload is bandwidth bound
// (~30 GFLOP vs ~1.5 GB traffic -> ~70us at 23.3 TB/s), so everything moves
// in 16-byte vectors: b128 LDS fragment loads (per ISA 7.12.2 layouts),
// async global<->LDS b128 staging (ASYNCcnt), coalesced b128 stores.
// ---------------------------------------------------------------------------

typedef __attribute__((ext_vector_type(16))) _Float16 v16h;
typedef __attribute__((ext_vector_type(8)))  _Float16 v8h;   // 16 bytes
typedef __attribute__((ext_vector_type(8)))  float    v8f;

#define TB 256      // threads per block (8 wave32)
#define MT 128      // M tile (8 waves x 16 rows)
#define NT 16       // N tile

// Low 32 bits of a flat pointer into LDS == LDS byte offset (generic AS3
// pointers are {aperture_hi32, lds_offset32}).
__device__ __forceinline__ unsigned ldsAddr(const void* p){
  return (unsigned)(unsigned long long)p;
}

// ---------------------------- utility kernels ------------------------------

__global__ void k_transpose_xyz(const float* __restrict__ in, float* __restrict__ out,
                                int B, int N){
  int i = blockIdx.x*blockDim.x + threadIdx.x;
  if (i >= B*N) return;
  int b = i / N, n = i % N;
  out[(size_t)i*3+0] = in[((size_t)b*3+0)*N + n];
  out[(size_t)i*3+1] = in[((size_t)b*3+1)*N + n];
  out[(size_t)i*3+2] = in[((size_t)b*3+2)*N + n];
}

// Farthest point sampling: one block per batch, dist in LDS, argmax tree
// reduction with lowest-index tie-break (matches jnp.argmax).
__global__ void k_fps(const float* __restrict__ xyz, int* __restrict__ fpsIdx,
                      float* __restrict__ newXyz, int N, int S){
  __shared__ float dist[4096];
  __shared__ float rmax[TB];
  __shared__ int   rarg[TB];
  __shared__ int   sfar;
  const int b = blockIdx.x, t = threadIdx.x;
  const float* P = xyz + (size_t)b*N*3;
  for (int j=t; j<N; j+=TB) dist[j] = 1e10f;
  if (t==0) sfar = 0;
  __syncthreads();
  for (int i=0;i<S;++i){
    int far = sfar;
    if (t==0){
      fpsIdx[b*S+i] = far;
      newXyz[((size_t)b*S+i)*3+0] = P[(size_t)far*3+0];
      newXyz[((size_t)b*S+i)*3+1] = P[(size_t)far*3+1];
      newXyz[((size_t)b*S+i)*3+2] = P[(size_t)far*3+2];
    }
    float cx=P[(size_t)far*3+0], cy=P[(size_t)far*3+1], cz=P[(size_t)far*3+2];
    float best=-1.f; int bidx=0;
    for (int j=t;j<N;j+=TB){
      float dx=P[(size_t)j*3]-cx, dy=P[(size_t)j*3+1]-cy, dz=P[(size_t)j*3+2]-cz;
      float d = dx*dx+dy*dy+dz*dz;
      float nd = fminf(dist[j], d);
      dist[j] = nd;
      if (nd > best){ best = nd; bidx = j; }
    }
    rmax[t]=best; rarg[t]=bidx;
    __syncthreads();
    for (int s=TB/2;s>0;s>>=1){
      if (t<s){
        if (rmax[t+s]>rmax[t] || (rmax[t+s]==rmax[t] && rarg[t+s]<rarg[t])){
          rmax[t]=rmax[t+s]; rarg[t]=rarg[t+s];
        }
      }
      __syncthreads();
    }
    if (t==0) sfar = rarg[0];
    __syncthreads();
  }
}

// Ball query, one wave32 per center: ballot + prefix-popcount compaction
// (equivalent to reference's sort-and-truncate of in-radius indices).
__global__ void k_query_ball(const float* __restrict__ xyz, const float* __restrict__ nxyz,
                             int* __restrict__ gidx, int Bn, int N, int S, int K, float r2){
  int w    = (blockIdx.x*blockDim.x + threadIdx.x) >> 5;
  int lane = threadIdx.x & 31;
  if (w >= Bn*S) return;
  int b = w / S;
  const float* P = xyz + (size_t)b*N*3;
  float cx=nxyz[(size_t)w*3], cy=nxyz[(size_t)w*3+1], cz=nxyz[(size_t)w*3+2];
  int* out = gidx + (size_t)w*K;
  int count=0, first=0; bool haveFirst=false;
  for (int j0=0; j0<N; j0+=32){
    int j = j0 + lane;
    bool inb = false;
    if (j < N){
      float dx=P[(size_t)j*3]-cx, dy=P[(size_t)j*3+1]-cy, dz=P[(size_t)j*3+2]-cz;
      inb = (dx*dx+dy*dy+dz*dz) <= r2;
    }
    unsigned mask = (unsigned)__ballot(inb);   // wave32: low 32 bits
    if (inb){
      int pos = count + __popc(mask & ((1u<<lane)-1u));
      if (pos < K) out[pos] = j;
    }
    if (!haveFirst && mask){ first = j0 + (__ffs((int)mask)-1); haveFirst = true; }
    count += __popc(mask);
    if (count >= K) break;
  }
  if (count < K){
    for (int p=count+lane; p<K; p+=32) out[p] = first;
  }
}

// --------------------------------- GEMM ------------------------------------
// Y[M,Co] = X[M,Ci] @ W[Co,Ci]^T + b.  mode 0: dense f16 X (leading dim ldx,
// zero-filled beyond Ci).  mode 1: fused gather of grouped input.

struct GemmArgs {
  const _Float16* X; const float* W; const float* bias; _Float16* Y;
  int M, Ci, Co, ldx;
  int mode;
  const int* gidx;
  const _Float16* pts; int Cpts;
  const float* xyz; const float* nxyz;
  int S, K, N;
};

__device__ __forceinline__ float fetchX(const GemmArgs& g, int r, int c){
  if (r >= g.M || c >= g.Ci) return 0.f;
  if (g.mode == 0) return (float)g.X[(size_t)r*g.ldx + c];
  int bs = r / g.K;
  int s  = bs % g.S;
  int b  = bs / g.S;
  int idx = g.gidx[r];
  if (c < g.Cpts) return (float)g.pts[((size_t)b*g.N + idx)*g.Cpts + c];
  int cc = c - g.Cpts; // 0..2 relative xyz
  return g.xyz[((size_t)b*g.N + idx)*3 + cc] - g.nxyz[((size_t)b*g.S + s)*3 + cc];
}

template<int KTILE>
__global__ void k_gemm(GemmArgs g){
  constexpr int LDAx = KTILE + 8;                 // halves; *2B = mult of 16B
  __shared__ _Float16 As[MT*LDAx];                // X tile, row-major
  __shared__ _Float16 Bs[KTILE*NT];               // W tile, K-MAJOR: Bs[k][n]
  const int t    = threadIdx.x;
  const int lane = t & 31;
  const int wave = t >> 5;
  const int half = lane >> 4;
  const int mBase = blockIdx.x * MT;
  const int nBase = blockIdx.y * NT;

  v8f acc = {};
  const int kTiles = (g.Ci + KTILE - 1) / KTILE;
  for (int kt=0; kt<kTiles; ++kt){
    const int k0 = kt*KTILE;
    // ---- stage A: MT*KTILE halves, 16-byte chunks ----
    #pragma unroll
    for (int e=0; e<(MT*KTILE/8)/TB; ++e){
      int chunk = e*TB + t;
      int i  = chunk / (KTILE/8);
      int j0 = (chunk % (KTILE/8)) * 8;
      int r = mBase + i, c0 = k0 + j0;
      if (g.mode == 0 && r < g.M && c0 + 8 <= g.ldx){
        // async global->LDS b128 (ASYNCcnt-tracked), no VGPR round-trip
        asm volatile("global_load_async_to_lds_b128 %0, %1, off"
                     :: "v"(ldsAddr(&As[i*LDAx + j0])),
                        "v"(&g.X[(size_t)r*g.ldx + c0])
                     : "memory");
        if (c0 + KTILE < g.ldx)                    // prefetch next K tile
          __builtin_prefetch(&g.X[(size_t)r*g.ldx + c0 + KTILE], 0, 1);
      } else {
        v8h val;
        #pragma unroll
        for (int q=0;q<8;++q) val[q] = (_Float16)fetchX(g, r, c0+q);
        *(v8h*)&As[i*LDAx + j0] = val;
      }
    }
    // ---- stage B (k-major): KTILE*16 halves ----
    #pragma unroll
    for (int e=0; e<(KTILE*NT)/TB; ++e){
      int idx = e*TB + t;
      int n = idx & 15, k = idx >> 4;
      int co = nBase + n, c = k0 + k;
      Bs[k*NT + n] = (co < g.Co && c < g.Ci) ? (_Float16)g.W[(size_t)co*g.Ci + c]
                                             : (_Float16)0.f;
    }
    asm volatile("s_wait_asynccnt 0" ::: "memory"); // async A-chunks done
    __syncthreads();
    // ---- fragments per ISA 7.12.2 (wave32): A: two b128 LDS loads, B: 16 strided 16-bit loads ----
    const int m = wave*16 + (lane & 15);
    #pragma unroll
    for (int kk0=0; kk0<KTILE; kk0+=32){
      v8h a0 = *(const v8h*)&As[m*LDAx + kk0 + 8*half];
      v8h a1 = *(const v8h*)&As[m*LDAx + kk0 + 16 + 8*half];
      v16h a = __builtin_shufflevector(a0, a1, 0,1,2,3,4,5,6,7,8,9,10,11,12,13,14,15);
      v8h b0, b1;
      #pragma unroll
      for (int j=0;j<8;++j){
        b0[j] = Bs[(kk0 + 8*half + j)*NT + (lane & 15)];
        b1[j] = Bs[(kk0 + 16 + 8*half + j)*NT + (lane & 15)];
      }
      v16h bfrag = __builtin_shufflevector(b0, b1, 0,1,2,3,4,5,6,7,8,9,10,11,12,13,14,15);
      acc = __builtin_amdgcn_wmma_f32_16x16x32_f16(
          false, a, false, bfrag, (short)0, acc, false, false);
    }
    __syncthreads();
  }
  // ---- epilogue: bias, bounce through LDS, async b128 LDS->global stores ----
  _Float16* Cs = As;                                // reuse (128*16 <= MT*LDAx)
  const int n  = lane & 15;
  float bv = g.bias[nBase + n];                     // Co is always a mult of 16
  #pragma unroll
  for (int v=0; v<8; ++v){
    int ml = wave*16 + v + 8*half;                  // C/D layout: m=v+8*(l>>4)
    Cs[ml*NT + n] = (_Float16)(acc[v] + bv);
  }
  __syncthreads();
  {
    int mrow = t >> 1, n0 = (t & 1) * 8;            // 128 rows x 2 chunks
    int m = mBase + mrow;
    if (m < g.M){
      asm volatile("global_store_async_from_lds_b128 %0, %1, off"
                   :: "v"(&g.Y[(size_t)m*g.Co + nBase + n0]),
                      "v"(ldsAddr(&Cs[mrow*NT + n0]))
                   : "memory");
    }
    asm volatile("s_wait_asynccnt 0" ::: "memory");
  }
}

// Deterministic per-channel sum/sumsq, 8 channels per block, b128 loads.
__global__ void k_stats(const _Float16* __restrict__ Y, float* __restrict__ stats,
                        int M, int Co){
  __shared__ float red[TB*8];
  const int c0 = blockIdx.x*8, t = threadIdx.x;
  float s[8], q[8];
  #pragma unroll
  for (int e=0;e<8;++e){ s[e]=0.f; q[e]=0.f; }
  for (int m=t; m<M; m+=TB){
    v8h y = *(const v8h*)&Y[(size_t)m*Co + c0];
    #pragma unroll
    for (int e=0;e<8;++e){ float v=(float)y[e]; s[e]+=v; q[e]+=v*v; }
  }
  #pragma unroll
  for (int e=0;e<8;++e) red[t*8+e]=s[e];
  __syncthreads();
  for (int st=TB/2; st>0; st>>=1){
    if (t<st){
      #pragma unroll
      for (int e=0;e<8;++e) red[t*8+e]+=red[(t+st)*8+e];
    }
    __syncthreads();
  }
  if (t<8) stats[c0+t]=red[t];
  __syncthreads();
  #pragma unroll
  for (int e=0;e<8;++e) red[t*8+e]=q[e];
  __syncthreads();
  for (int st=TB/2; st>0; st>>=1){
    if (t<st){
      #pragma unroll
      for (int e=0;e<8;++e) red[t*8+e]+=red[(t+st)*8+e];
    }
    __syncthreads();
  }
  if (t<8) stats[Co+c0+t]=red[t];
}

__global__ void k_bn_finalize(float* stats, const float* __restrict__ gamma,
                              const float* __restrict__ beta, int Co, float invM){
  int c = blockIdx.x*blockDim.x + threadIdx.x;
  if (c >= Co) return;
  float mean = stats[c]*invM;
  float var  = stats[Co+c]*invM - mean*mean;
  float sc   = gamma[c] * rsqrtf(var + 1e-5f);
  stats[c]    = sc;                  // scale
  stats[Co+c] = beta[c] - mean*sc;   // shift
}

__global__ void k_bn_relu(_Float16* Y, const float* __restrict__ ss, int M, int Co,
                          float* __restrict__ outF){
  int i = blockIdx.x*blockDim.x + threadIdx.x;     // 8-element chunk id
  if (i >= (M*Co)/8) return;
  int c0 = (i*8) % Co;                             // Co % 8 == 0 always
  v8h y = *(const v8h*)&Y[(size_t)i*8];
  v8h r;
  #pragma unroll
  for (int e=0;e<8;++e){
    float v = (float)y[e]*ss[c0+e] + ss[Co+c0+e];
    v = fmaxf(v, 0.f);
    r[e] = (_Float16)v;
    if (outF) outF[(size_t)i*8+e] = v;
  }
  *(v8h*)&Y[(size_t)i*8] = r;
}

__global__ void k_maxpool(const _Float16* __restrict__ Y, int G, int K, int C,
                          _Float16* __restrict__ outH, int ostride, int coff,
                          float* __restrict__ outF){
  int C8 = C/8;
  int i = blockIdx.x*blockDim.x + threadIdx.x;
  if (i >= G*C8) return;
  int gI = i / C8, c0 = (i % C8)*8;
  float m[8];
  #pragma unroll
  for (int e=0;e<8;++e) m[e] = -3.4e38f;
  const _Float16* p = Y + ((size_t)gI*K)*C + c0;
  for (int k=0;k<K;++k){
    v8h y = *(const v8h*)&p[(size_t)k*C];
    #pragma unroll
    for (int e=0;e<8;++e) m[e] = fmaxf(m[e], (float)y[e]);
  }
  if (outH){
    v8h r;
    #pragma unroll
    for (int e=0;e<8;++e) r[e]=(_Float16)m[e];
    *(v8h*)&outH[(size_t)gI*ostride + coff + c0] = r;
  }
  if (outF){
    #pragma unroll
    for (int e=0;e<8;++e) outF[(size_t)gI*C + c0 + e] = m[e];
  }
}

__global__ void k_concat_sa3(const float* __restrict__ l2xyz,
                             const _Float16* __restrict__ l2pts,
                             _Float16* __restrict__ X, int rows, int Cpts, int ld){
  int C = 3 + Cpts;
  int i = blockIdx.x*blockDim.x + threadIdx.x;
  if (i >= rows*ld) return;
  int r = i / ld, c = i % ld;
  _Float16 v = (_Float16)0.f;
  if (c < 3)      v = (_Float16)l2xyz[(size_t)r*3 + c];
  else if (c < C) v = l2pts[(size_t)r*Cpts + (c-3)];
  X[i] = v;
}

// ------------------------------- host side ---------------------------------

struct DenseP { const float *W,*b,*gamma,*beta; };

static DenseP takeDense(void* const* din, int& i, bool gammaFirst){
  DenseP d;
  d.W = (const float*)din[i++];
  d.b = (const float*)din[i++];
  if (gammaFirst){ d.gamma=(const float*)din[i++]; d.beta=(const float*)din[i++]; }
  else           { d.beta =(const float*)din[i++]; d.gamma=(const float*)din[i++]; }
  return d;
}

extern "C" void kernel_launch(void* const* d_in, const int* in_sizes, int n_in,
                              void* d_out, int out_size, void* d_ws, size_t ws_size,
                              hipStream_t stream) {
  (void)n_in; (void)out_size; (void)ws_size;
  const int B = 16, N1 = 4096, S1 = 512, S2 = 128;

  // ---- parameter pointer mapping (handles both flatten conventions) ----
  const float* xyz_in = nullptr;
  DenseP sa1p[3][3], sa2p[3][3], sa3p[3], fc1p, fc2p;
  if (in_sizes[0] == B*3*N1){
    int i = 0;                       // insertion order; W,b,gamma,beta
    xyz_in = (const float*)d_in[i++];
    for (int br=0;br<3;++br) for (int l=0;l<3;++l) sa1p[br][l] = takeDense(d_in,i,true);
    for (int br=0;br<3;++br) for (int l=0;l<3;++l) sa2p[br][l] = takeDense(d_in,i,true);
    for (int l=0;l<3;++l) sa3p[l] = takeDense(d_in,i,true);
    fc1p = takeDense(d_in,i,true);
    fc2p = takeDense(d_in,i,true);
  } else {
    int i = 0;                       // sorted pytree order; W,b,beta,gamma
    fc1p = takeDense(d_in,i,false);
    fc2p = takeDense(d_in,i,false);
    for (int br=0;br<3;++br) for (int l=0;l<3;++l) sa1p[br][l] = takeDense(d_in,i,false);
    for (int br=0;br<3;++br) for (int l=0;l<3;++l) sa2p[br][l] = takeDense(d_in,i,false);
    for (int l=0;l<3;++l) sa3p[l] = takeDense(d_in,i,false);
    xyz_in = (const float*)d_in[i++];
  }

  // ---- workspace carve ----
  char* wp = (char*)d_ws;
  auto alloc = [&](size_t bytes)->void*{
    void* r = (void*)wp; wp += (bytes + 255) & ~(size_t)255; return r;
  };
  float*    xyzT  = (float*)   alloc((size_t)B*N1*3*4);
  float*    nx1   = (float*)   alloc((size_t)B*S1*3*4);
  int*      fps1  = (int*)     alloc((size_t)B*S1*4);
  float*    nx2   = (float*)   alloc((size_t)B*S2*3*4);
  int*      fps2  = (int*)     alloc((size_t)B*S2*4);
  int*      gidx  = (int*)     alloc((size_t)B*S1*128*4);
  _Float16* bufA  = (_Float16*)alloc((size_t)B*S1*128*64*2);
  _Float16* bufB  = (_Float16*)alloc((size_t)B*S1*128*64*2);
  _Float16* l1pts = (_Float16*)alloc((size_t)B*S1*160*2);
  _Float16* l2pts = (_Float16*)alloc((size_t)B*S2*320*2);
  _Float16* x3    = (_Float16*)alloc((size_t)B*S2*328*2);  // ld 328 (8-aligned)
  _Float16* l3h   = (_Float16*)alloc((size_t)B*512*2);
  float*    stats = (float*)   alloc((size_t)2*512*4);

  float* outF = (float*)d_out;

  // ---- GEMM + BN + ReLU layer driver ----
  auto runDense = [&](const DenseP& L, GemmArgs g, float* bnOutF){
    g.W = L.W; g.bias = L.b;
    dim3 grid((g.M + MT - 1)/MT, (g.Co + NT - 1)/NT);
    if (g.Ci >= 64) k_gemm<64><<<grid, dim3(TB), 0, stream>>>(g);
    else            k_gemm<32><<<grid, dim3(TB), 0, stream>>>(g);
    k_stats<<<dim3(g.Co/8), dim3(TB), 0, stream>>>(g.Y, stats, g.M, g.Co);
    k_bn_finalize<<<dim3((g.Co+255)/256), dim3(256), 0, stream>>>(
        stats, L.gamma, L.beta, g.Co, 1.0f/(float)g.M);
    int chunks = (g.M * g.Co) / 8;
    k_bn_relu<<<dim3((chunks+255)/256), dim3(256), 0, stream>>>(g.Y, stats, g.M, g.Co, bnOutF);
  };

  // ---- one MSG branch: ball query -> 3 layers -> maxpool into concat buf ----
  auto runBranch = [&](const DenseP* layers, const float* bxyz, int N,
                       const float* bnx, int S, const _Float16* pts, int Cpts,
                       float radius, int K, const int* ch,
                       _Float16* outPts, int ostride, int coff){
    int warps = B*S;
    k_query_ball<<<dim3((warps*32 + TB - 1)/TB), dim3(TB), 0, stream>>>(
        bxyz, bnx, gidx, B, N, S, K, radius*radius);
    int M = B*S*K;
    GemmArgs g{};                       // layer 0: fused gather
    g.Y = bufA; g.M = M; g.Ci = Cpts + 3; g.Co = ch[0];
    g.mode = 1; g.gidx = gidx; g.pts = pts; g.Cpts = Cpts;
    g.xyz = bxyz; g.nxyz = bnx; g.S = S; g.K = K; g.N = N;
    runDense(layers[0], g, nullptr);
    GemmArgs h{};                       // layer 1: dense
    h.mode = 0; h.X = bufA; h.Y = bufB; h.M = M; h.Ci = ch[0]; h.ldx = ch[0]; h.Co = ch[1];
    runDense(layers[1], h, nullptr);
    h.X = bufB; h.Y = bufA; h.Ci = ch[1]; h.ldx = ch[1]; h.Co = ch[2];   // layer 2
    runDense(layers[2], h, nullptr);
    int tot = B*S*(ch[2]/8);
    k_maxpool<<<dim3((tot+255)/256), dim3(256), 0, stream>>>(
        bufA, B*S, K, ch[2], outPts, ostride, coff, nullptr);
  };

  // ---- stage 0: layout + FPS ----
  k_transpose_xyz<<<dim3((B*N1+255)/256), dim3(256), 0, stream>>>(xyz_in, xyzT, B, N1);
  k_fps<<<dim3(B), dim3(TB), 0, stream>>>(xyzT, fps1, nx1, N1, S1);

  // ---- sa1 (points = None) ----
  { int ch[3]={16,16,32}; runBranch(sa1p[0], xyzT, N1, nx1, S1, nullptr, 0, 0.1f,  16, ch, l1pts, 160,  0); }
  { int ch[3]={32,32,64}; runBranch(sa1p[1], xyzT, N1, nx1, S1, nullptr, 0, 0.2f,  32, ch, l1pts, 160, 32); }
  { int ch[3]={32,32,64}; runBranch(sa1p[2], xyzT, N1, nx1, S1, nullptr, 0, 0.4f, 128, ch, l1pts, 160, 96); }

  // ---- sa2 (points = l1pts, N=512) ----
  k_fps<<<dim3(B), dim3(TB), 0, stream>>>(nx1, fps2, nx2, S1, S2);
  { int ch[3]={32,32,64};   runBranch(sa2p[0], nx1, S1, nx2, S2, l1pts, 160, 0.2f,  32, ch, l2pts, 320,   0); }
  { int ch[3]={64,64,128};  runBranch(sa2p[1], nx1, S1, nx2, S2, l1pts, 160, 0.4f,  64, ch, l2pts, 320,  64); }
  { int ch[3]={64,64,128};  runBranch(sa2p[2], nx1, S1, nx2, S2, l1pts, 160, 0.8f, 128, ch, l2pts, 320, 192); }

  // ---- sa3: pointwise MLP on concat(l2_xyz, l2_pts), then global maxpool ----
  {
    int rows = B*S2; // 2048
    k_concat_sa3<<<dim3((rows*328+255)/256), dim3(256), 0, stream>>>(nx2, l2pts, x3, rows, 320, 328);
    GemmArgs g{}; g.mode = 0;
    g.X = x3;   g.Y = bufA; g.M = rows; g.Ci = 323; g.ldx = 328; g.Co = 128; runDense(sa3p[0], g, nullptr);
    g.X = bufA; g.Y = bufB; g.Ci = 128; g.ldx = 128; g.Co = 256;             runDense(sa3p[1], g, nullptr);
    g.X = bufB; g.Y = bufA; g.Ci = 256; g.ldx = 256; g.Co = 512;             runDense(sa3p[2], g, nullptr);
    // l3 = max over the 128 points; f32 copy straight into d_out[2048:10240]
    k_maxpool<<<dim3((B*512/8+255)/256), dim3(256), 0, stream>>>(
        bufA, B, S2, 512, l3h, 512, 0, outF + 2048);
    // fc1 / fc2 heads (M = 16, single WMMA M-tile)
    GemmArgs f{}; f.mode = 0;
    f.X = l3h;  f.Y = bufB; f.M = B; f.Ci = 512; f.ldx = 512; f.Co = 256; runDense(fc1p, f, nullptr);
    f.X = bufB; f.Y = bufA; f.Ci = 256; f.ldx = 256; f.Co = 128;          runDense(fc2p, f, outF);
  }
}
